// MoNet_26817775796897
// MI455X (gfx1250) — matrix operand order repaired and checked by reading, not compiled, not run on hardware
//
#include <hip/hip_runtime.h>
#include <hip/hip_bf16.h>
#include <math.h>

// ---------------------------------------------------------------------------
// MoNet / GMMConv pipeline for gfx1250 (MI455X).
//  - Dense node GEMMs run on v_wmma_f32_16x16x32_bf16 with operands pre-packed
//    into bf16 fragment-order buffers (one contiguous 32B load per lane
//    fragment, zero guards / converts in the hot loop).
//  - Edge pass: per-edge Gaussian weights precomputed, then gather xw[src]
//    (L2-resident) + fp32 atomic scatter into agg[dst].
//  - Degree is layer-invariant -> computed once.
// ---------------------------------------------------------------------------

typedef __attribute__((ext_vector_type(16))) __bf16 v16bf;
typedef __attribute__((ext_vector_type(8)))  float  v8f;

#define KKER 3      // GMM kernel count
#define WPB  4      // waves per block in GEMM

// ---------------------------------------------------------------------------
__global__ void fill_zero_kernel(float* __restrict__ p, long long n) {
    long long i = (long long)blockIdx.x * blockDim.x + threadIdx.x;
    if (i < n) p[i] = 0.0f;
}

__global__ void degree_kernel(const int* __restrict__ dst, float* __restrict__ deg,
                              int E) {
    int e = blockIdx.x * blockDim.x + threadIdx.x;
    if (e < E) atomicAdd(&deg[dst[e]], 1.0f);
}

// ---------------------------------------------------------------------------
// Pack A (node features, f32 [Mrows,Ka]) -> bf16 [Mpad,K32], zero padded,
// each 32-K block kept in contiguous k order (k 0..31), matching the
// B fragment K range (hsel*16 + 0..15).
// A lane then reads 16 contiguous bf16 (= its whole fragment) at
// row*K32 + kb + hsel*16.
// ---------------------------------------------------------------------------
__global__ void pack_a_kernel(const float* __restrict__ A, __bf16* __restrict__ Ap,
                              int Mrows, int Mpad, int Ka, int K32) {
    long long tid = (long long)blockIdx.x * blockDim.x + threadIdx.x;
    long long total = (long long)Mpad * K32;
    if (tid >= total) return;
    int row = (int)(tid / K32);
    int d   = (int)(tid % K32);
    int blk = d & ~31;
    int db  = d & 31;
    int g   = db >> 3;                         // dest group 0..3
    int sg  = g;                               // source group
    int k   = blk + sg * 8 + (db & 7);
    float v = (row < Mrows && k < Ka) ? A[(long long)row * Ka + k] : 0.0f;
    Ap[tid] = (__bf16)v;
}

// ---------------------------------------------------------------------------
// Pack B (weights, f32 [Ka,Nn]) -> bf16 column-major [Nn,K32], zero padded.
// B fragment K range (hsel*16 + 0..15) is naturally contiguous here.
// ---------------------------------------------------------------------------
__global__ void pack_b_kernel(const float* __restrict__ B, __bf16* __restrict__ Bt,
                              int Ka, int K32, int Nn) {
    int tid = blockIdx.x * blockDim.x + threadIdx.x;
    int total = Nn * K32;
    if (tid >= total) return;
    int col = tid / K32;
    int k   = tid % K32;
    float v = (k < Ka) ? B[(long long)k * Nn + col] : 0.0f;
    Bt[tid] = (__bf16)v;
}

// ---------------------------------------------------------------------------
// WMMA GEMM: C[Mreal,Nn] = Ap * Bt  (bf16 in, f32 accumulate/out).
// One wave per 16x16 M-tile; each wave produces NT N-tiles, reusing the A
// fragment across the NT wmmas. grid.x * NT must equal Nn/16 (no col guards).
// ---------------------------------------------------------------------------
template <int NT>
__global__ void gemm_bf16_wmma_kernel(const __bf16* __restrict__ Ap, // [Mpad,K32]
                                      const __bf16* __restrict__ Bt, // [Nn,K32]
                                      float* __restrict__ C,         // [Mreal,Nn]
                                      int Mreal, int Mtiles, int K32, int Nn) {
    const int lane = threadIdx.x & 31;
    const int wave = threadIdx.x >> 5;
    const int hsel = lane >> 4;
    const int l16  = lane & 15;

    const int mtile = blockIdx.y * WPB + wave;
    if (mtile >= Mtiles) return;

    const int row    = mtile * 16 + l16;
    const int ntile0 = blockIdx.x * NT;

    const v16bf* arow = (const v16bf*)(Ap + (long long)row * K32);

    v8f acc[NT];
#pragma unroll
    for (int t = 0; t < NT; ++t) acc[t] = (v8f){};

    const int nblk = K32 >> 5;
    for (int bi = 0; bi < nblk; ++bi) {
        v16bf a = arow[bi * 2 + hsel];
#pragma unroll
        for (int t = 0; t < NT; ++t) {
            int col = (ntile0 + t) * 16 + l16;
            const v16bf* bcol = (const v16bf*)(Bt + (long long)col * K32);
            v16bf b = bcol[bi * 2 + hsel];
            acc[t] = __builtin_amdgcn_wmma_f32_16x16x32_bf16(
                         false, a, false, b, (short)0, acc[t], false, false);
        }
    }

#pragma unroll
    for (int t = 0; t < NT; ++t) {
        int col = (ntile0 + t) * 16 + l16;
#pragma unroll
        for (int r = 0; r < 8; ++r) {
            int ro = mtile * 16 + r + hsel * 8;   // D: VGPR r -> row r + 8*half
            if (ro < Mreal) C[(long long)ro * Nn + col] = acc[t][r];
        }
    }
}

// ---------------------------------------------------------------------------
// Per-edge Gaussian kernel weights: wgt[e] = (w0,w1,w2,0)
// ---------------------------------------------------------------------------
__global__ void gauss_kernel(const float* __restrict__ pseudo, // [E,2]
                             const float* __restrict__ mu,     // [3,2]
                             const float* __restrict__ sigma,  // [3,2]
                             float4* __restrict__ wgt,         // [E]
                             int E) {
    int e = blockIdx.x * blockDim.x + threadIdx.x;
    if (e >= E) return;
    float p0 = pseudo[2 * e + 0];
    float p1 = pseudo[2 * e + 1];
    float w[KKER];
#pragma unroll
    for (int k = 0; k < KKER; ++k) {
        float d0 = p0 - mu[2 * k + 0];
        float d1 = p1 - mu[2 * k + 1];
        float s0 = sigma[2 * k + 0];
        float s1 = sigma[2 * k + 1];
        w[k] = __expf(-0.5f * (d0 * d0 / (1e-15f + s0 * s0) +
                               d1 * d1 / (1e-15f + s1 * s1)));
    }
    wgt[e] = make_float4(w[0], w[1], w[2], 0.0f);
}

// ---------------------------------------------------------------------------
// Edge gather + scatter: thread = 4 channels of one edge.
//   agg[dst] += sum_k w_k * xw[src, k*Cout + c..c+3]
// ---------------------------------------------------------------------------
__global__ void edge_scatter_kernel(const int*    __restrict__ src,
                                    const int*    __restrict__ dst,
                                    const float4* __restrict__ wgt, // [E]
                                    const float*  __restrict__ xw,  // [N,3*Cout]
                                    float*        __restrict__ agg, // [N,Cout]
                                    int E, int Cout) {
    const int cpt = Cout >> 2;
    long long tid = (long long)blockIdx.x * blockDim.x + threadIdx.x;
    int e  = (int)(tid / cpt);
    int c4 = (int)(tid % cpt) * 4;
    if (e >= E) return;

    const int s = src[e];
    const int d = dst[e];
    const float4 w = wgt[e];

    const float4* xrow = (const float4*)(xw + (long long)s * (KKER * Cout));
    float4 m = make_float4(0.f, 0.f, 0.f, 0.f);
    float4 v0 = xrow[(0 * Cout + c4) >> 2];
    float4 v1 = xrow[(1 * Cout + c4) >> 2];
    float4 v2 = xrow[(2 * Cout + c4) >> 2];
    m.x = w.x * v0.x + w.y * v1.x + w.z * v2.x;
    m.y = w.x * v0.y + w.y * v1.y + w.z * v2.y;
    m.z = w.x * v0.z + w.y * v1.z + w.z * v2.z;
    m.w = w.x * v0.w + w.y * v1.w + w.z * v2.w;

    float* ap = agg + (long long)d * Cout + c4;
    atomicAdd(ap + 0, m.x);
    atomicAdd(ap + 1, m.y);
    atomicAdd(ap + 2, m.z);
    atomicAdd(ap + 3, m.w);
}

// ---------------------------------------------------------------------------
__global__ void finalize_kernel(const float* __restrict__ agg,
                                const float* __restrict__ deg,
                                const float* __restrict__ rootout,
                                const float* __restrict__ bias,
                                float* __restrict__ hout,
                                int Nnodes, int Cout) {
    long long tid = (long long)blockIdx.x * blockDim.x + threadIdx.x;
    long long total = (long long)Nnodes * Cout;
    if (tid >= total) return;
    int n = (int)(tid / Cout);
    int c = (int)(tid % Cout);
    float dg = deg[n];
    dg = dg < 1.0f ? 1.0f : dg;
    float v = agg[tid] / dg + rootout[tid] + bias[c];
    hout[tid] = v > 0.0f ? v : 0.0f;
}

__global__ void head_kernel(const float* __restrict__ h,    // [N,64]
                            const float* __restrict__ fcw,  // [64,2]
                            const float* __restrict__ fcb,  // [2]
                            float* __restrict__ out,        // [N,2]
                            int Nnodes, int Cin) {
    int n = blockIdx.x * blockDim.x + threadIdx.x;
    if (n >= Nnodes) return;
    float z0 = fcb[0], z1 = fcb[1];
    const float* hr = h + (long long)n * Cin;
    for (int c = 0; c < Cin; ++c) {
        float v = hr[c];
        z0 += v * fcw[2 * c + 0];
        z1 += v * fcw[2 * c + 1];
    }
    float mx  = fmaxf(z0, z1);
    float lse = mx + __logf(__expf(z0 - mx) + __expf(z1 - mx));
    out[2 * n + 0] = z0 - lse;
    out[2 * n + 1] = z1 - lse;
}

// ---------------------------------------------------------------------------
static inline dim3 grid1d(long long total, int block) {
    return dim3((unsigned)((total + block - 1) / block));
}

static void launch_gemm(const __bf16* Ap, const __bf16* Bt, float* C,
                        int Mreal, int Mtiles, int K32, int Nn,
                        hipStream_t stream) {
    dim3 blk(32 * WPB);
    unsigned gy = (unsigned)((Mtiles + WPB - 1) / WPB);
    int ntiles = Nn / 16;
    if (ntiles % 6 == 0) {
        dim3 g(ntiles / 6, gy);
        gemm_bf16_wmma_kernel<6><<<g, blk, 0, stream>>>(Ap, Bt, C, Mreal, Mtiles, K32, Nn);
    } else {
        dim3 g(ntiles / 2, gy);
        gemm_bf16_wmma_kernel<2><<<g, blk, 0, stream>>>(Ap, Bt, C, Mreal, Mtiles, K32, Nn);
    }
}

extern "C" void kernel_launch(void* const* d_in, const int* in_sizes, int n_in,
                              void* d_out, int out_size, void* d_ws, size_t ws_size,
                              hipStream_t stream) {
    (void)n_in; (void)ws_size; (void)out_size;

    const float* x     = (const float*)d_in[0];
    const int*   eidx  = (const int*)  d_in[1];
    const float* eattr = (const float*)d_in[2];
    const float* fc_w  = (const float*)d_in[18];
    const float* fc_b  = (const float*)d_in[19];

    const int NUM_FEATURES = 22;
    const int N = in_sizes[0] / NUM_FEATURES;   // 163842
    const int E = in_sizes[1] / 2;              // 983040
    const int* src = eidx;
    const int* dst = eidx + E;

    const int Mtiles = (N + 15) / 16;
    const int Mpad   = Mtiles * 16;

    struct Layer {
        const float *g, *mu, *sigma, *root, *bias;
        int cin, cout;
    };
    Layer L[3] = {
        { (const float*)d_in[3],  (const float*)d_in[4],  (const float*)d_in[5],
          (const float*)d_in[6],  (const float*)d_in[7],  22, 32 },
        { (const float*)d_in[8],  (const float*)d_in[9],  (const float*)d_in[10],
          (const float*)d_in[11], (const float*)d_in[12], 32, 64 },
        { (const float*)d_in[13], (const float*)d_in[14], (const float*)d_in[15],
          (const float*)d_in[16], (const float*)d_in[17], 64, 64 },
    };

    // Workspace layout (256B aligned regions)
    char*  base = (char*)d_ws;
    size_t off  = 0;
    auto carve = [&](size_t bytes) -> void* {
        void* p = (void*)(base + off);
        off += (bytes + 255) & ~(size_t)255;
        return p;
    };
    float*  deg     = (float*) carve((size_t)N * 4);
    float*  xw      = (float*) carve((size_t)N * KKER * 64 * 4);
    float*  rootout = (float*) carve((size_t)N * 64 * 4);
    float*  agg     = (float*) carve((size_t)N * 64 * 4);
    float*  hA      = (float*) carve((size_t)N * 64 * 4);
    float*  hB      = (float*) carve((size_t)N * 64 * 4);
    float4* wgt     = (float4*)carve((size_t)E * 16);
    __bf16* Ap      = (__bf16*)carve((size_t)Mpad * 64 * 2);
    __bf16* BtG     = (__bf16*)carve((size_t)192 * 64 * 2);  // packed g
    __bf16* BtR     = (__bf16*)carve((size_t)64  * 64 * 2);  // packed root

    const int BLK = 256;

    // Degree (layer-invariant)
    fill_zero_kernel<<<grid1d(N, BLK), BLK, 0, stream>>>(deg, N);
    degree_kernel<<<grid1d(E, BLK), BLK, 0, stream>>>(dst, deg, E);

    const float* hcur = x;
    float* hnext = hA;
    float* hother = hB;

    for (int li = 0; li < 3; ++li) {
        const Layer& P = L[li];
        const int cin  = P.cin;
        const int cout = P.cout;
        const int nxw  = KKER * cout;          // 96 / 192 / 192
        const int K32  = (cin + 31) & ~31;     // 32 / 32 / 64

        // zero agg
        fill_zero_kernel<<<grid1d((long long)N * cout, BLK), BLK, 0, stream>>>(
            agg, (long long)N * cout);

        // pack operands to bf16 fragment-order buffers
        pack_a_kernel<<<grid1d((long long)Mpad * K32, BLK), BLK, 0, stream>>>(
            hcur, Ap, N, Mpad, cin, K32);
        pack_b_kernel<<<grid1d(nxw * K32, BLK), BLK, 0, stream>>>(
            P.g, BtG, cin, K32, nxw);
        pack_b_kernel<<<grid1d(cout * K32, BLK), BLK, 0, stream>>>(
            P.root, BtR, cin, K32, cout);

        // xw = h @ g ; rootout = h @ root   (WMMA bf16)
        launch_gemm(Ap, BtG, xw,      N, Mtiles, K32, nxw,  stream);
        launch_gemm(Ap, BtR, rootout, N, Mtiles, K32, cout, stream);

        // per-edge Gaussian weights, then gather+scatter
        gauss_kernel<<<grid1d(E, BLK), BLK, 0, stream>>>(
            eattr, P.mu, P.sigma, wgt, E);
        {
            long long total = (long long)E * (cout >> 2);
            edge_scatter_kernel<<<grid1d(total, BLK), BLK, 0, stream>>>(
                src, dst, wgt, xw, agg, E, cout);
        }

        // h_next = relu(agg/deg + rootout + bias)
        {
            long long total = (long long)N * cout;
            finalize_kernel<<<grid1d(total, BLK), BLK, 0, stream>>>(
                agg, deg, rootout, P.bias, hnext, N, cout);
        }

        hcur = hnext;
        float* t = hnext; hnext = hother; hother = t;
    }

    // FC head + log_softmax -> d_out [N,2]
    head_kernel<<<grid1d(N, BLK), BLK, 0, stream>>>(
        hcur, fc_w, fc_b, (float*)d_out, N, 64);
}
